// VisionMamba_40046275068516
// MI455X (gfx1250) — matrix operand order repaired and checked by reading, not compiled, not run on hardware
//
#include <hip/hip_runtime.h>

// ---------------- CDNA5 types ----------------
typedef __attribute__((ext_vector_type(16))) __bf16 bf16x16;
typedef __attribute__((ext_vector_type(8)))  float  v8f;
typedef __attribute__((ext_vector_type(4)))  unsigned int u32x4;
typedef __attribute__((ext_vector_type(8)))  unsigned int u32x8;
typedef __attribute__((ext_vector_type(4)))  float f32x4;
typedef __attribute__((ext_vector_type(8)))  int i32x8;
typedef __attribute__((ext_vector_type(4)))  int i32x4;

#define DEPTH    8
#define D_MODEL  768
#define D_INNER  1536
#define D_XZ     3072
#define D_STATE  16
#define DT_RANK  48
#define DT_PAD   64
#define XDBL_N   80          // dt_rank + 2*d_state
#define LSEQ     301
#define NPATCH   300
#define MIDTOK   150
#define MROWS    602         // B * L
#define PK       12288       // 3*64*64
#define GW_      15
#define EPSF     1e-5f

struct H16 { __bf16 h[16]; };

__device__ __forceinline__ __bf16 f2b(float x) { return (__bf16)x; }
__device__ __forceinline__ float  silu(float x) { return x / (1.f + __expf(-x)); }

// Concatenate two 16B chunks into one 32B WMMA fragment.
__device__ __forceinline__ bf16x16 ld_pair(const __bf16* p0, const __bf16* p1) {
  u32x4 a = *(const u32x4*)p0;
  u32x4 b = *(const u32x4*)p1;
  u32x8 c = __builtin_shufflevector(a, b, 0, 1, 2, 3, 4, 5, 6, 7);
  return __builtin_bit_cast(bf16x16, c);
}

// A fragment (16x32 bf16, row-major activations, K contiguous).
__device__ __forceinline__ bf16x16 load_afrag(const __bf16* A, int lda, int row_base,
                                              int kbase, int lane, int maxRow) {
  int r = row_base + (lane & 15);
  r = (r > maxRow) ? maxRow : r;
  const __bf16* p = A + (size_t)r * lda + kbase + ((lane >> 4) << 3);
  return ld_pair(p, p + 16);
}

// B fragment (32x16 bf16) from weight stored (out,in) row-major = B^T.
__device__ __forceinline__ bf16x16 load_bfrag(const __bf16* Bt, int ldb, int n_base,
                                              int kbase, int lane) {
  int n = n_base + (lane & 15);
  const __bf16* p = Bt + (size_t)n * ldb + kbase + ((lane >> 4) << 3);
  return ld_pair(p, p + 16);
}

// B fragment from an LDS-resident 64x32 bf16 tile (row-major, rows = output cols).
__device__ __forceinline__ bf16x16 bfrag_from_lds(const __bf16* buf, int t, int lane) {
  int r = 16 * t + (lane & 15);
  const __bf16* p = buf + r * 32 + ((lane >> 4) << 3);
  return ld_pair(p, p + 16);
}

__device__ __forceinline__ v8f wmma_bf(bf16x16 a, bf16x16 b, v8f c) {
  return __builtin_amdgcn_wmma_f32_16x16x32_bf16(false, a, false, b, (short)0, c, false, false);
}

__device__ __forceinline__ void zero8(v8f& v) {
#pragma unroll
  for (int i = 0; i < 8; ++i) v[i] = 0.f;
}

// ---------------- Tensor Data Mover (TDM) ----------------
// D# built per CDNA5 ISA: 2D tile, 2-byte elements,
// tile_dim0=32 (K chunk), tile_dim1=64 (output-channel rows), contiguous LDS dest.
// This toolchain exposes the 6-arg builtin: (g0, g1, g2, g3, i32x8, cpol).
#if __has_builtin(__builtin_amdgcn_tensor_load_to_lds)
#define HAVE_TDM 1
__device__ __forceinline__ void tdm_load_64x32_bf16(unsigned lds_addr, const __bf16* g,
                                                    unsigned row_stride_elems) {
  unsigned long long ga = (unsigned long long)(uintptr_t)g;
  u32x4 g0;
  g0[0] = 1u;                                   // count=1, user descriptor
  g0[1] = lds_addr;                             // LDS byte address
  g0[2] = (unsigned)ga;                         // global_addr[31:0]
  g0[3] = (unsigned)((ga >> 32) & 0x01FFFFFFull) | (2u << 30);  // addr[56:32] | type=2
  i32x8 g1;
  g1[0] = 0x00010000;                           // data_size=1 (2 bytes), mask=0
  g1[1] = (int)((row_stride_elems & 0xFFFFu) << 16);        // tensor_dim0[15:0]
  g1[2] = (int)((row_stride_elems >> 16) & 0xFFFFu);        // tensor_dim0[31:16], dim1 lo=0
  g1[3] = (int)(0x10u | (32u << 16));           // tensor_dim1=1<<20 (hi), tile_dim0=32
  g1[4] = 64;                                   // tile_dim1=64, tile_dim2=0
  g1[5] = (int)row_stride_elems;                // tensor_dim0_stride[31:0]
  g1[6] = 0;                                    // stride0 hi, stride1 lo
  g1[7] = 0;                                    // stride1 hi
  i32x4 gz;
  gz[0] = 0; gz[1] = 0; gz[2] = 0; gz[3] = 0;   // groups 2/3 unused (2D)
  i32x8 gz8;
#pragma unroll
  for (int i = 0; i < 8; ++i) gz8[i] = 0;
  __builtin_amdgcn_tensor_load_to_lds(g0, g1, gz, gz, gz8, 0);
}
#else
#define HAVE_TDM 0
#endif

// ---------------- utility kernels ----------------
__global__ void k_zero_f32(float* __restrict__ p, long n) {
  long i = (long)blockIdx.x * blockDim.x + threadIdx.x;
  if (i < n) p[i] = 0.f;
}
__global__ void k_zero_bf16(__bf16* __restrict__ p, long n) {
  long i = (long)blockIdx.x * blockDim.x + threadIdx.x;
  if (i < n) p[i] = (__bf16)0.f;
}
__global__ void k_cvt(const float* __restrict__ s, __bf16* __restrict__ d, long n) {
  long i = (long)blockIdx.x * blockDim.x + threadIdx.x;
  if (i < n) d[i] = f2b(s[i]);
}
__global__ void k_cvt_pad(const float* __restrict__ s, __bf16* __restrict__ d,
                          int rows, int cin, int cpad) {
  long i = (long)blockIdx.x * blockDim.x + threadIdx.x;
  long total = (long)rows * cpad;
  if (i >= total) return;
  int r = (int)(i / cpad), c = (int)(i % cpad);
  d[i] = (c < cin) ? f2b(s[(long)r * cin + c]) : (__bf16)0.f;
}

// ---------------- patch embed GEMM (implicit im2col, TDM weight stream) ------------
__global__ __launch_bounds__(128) void k_patch_gemm(
    const float* __restrict__ x, const __bf16* __restrict__ Wt,
    const float* __restrict__ pbias, const float* __restrict__ pos,
    float* __restrict__ tokens) {
  int lane = threadIdx.x & 31, wave = threadIdx.x >> 5;
  int khalf = lane >> 4;
  int row_base = blockIdx.x * 16;
  int n_base0 = (blockIdx.y * 4 + wave) * 64;

  int r = row_base + (lane & 15); r = (r > 599) ? 599 : r;
  int b = r / NPATCH, p = r % NPATCH;
  int gh = p / GW_, gw = p % GW_;
  const float* xb = x + (size_t)b * 3 * 1280 * 960 + (size_t)(gh * 64) * 960 + gw * 64;

  v8f acc[4];
#pragma unroll
  for (int t = 0; t < 4; ++t) zero8(acc[t]);

#if HAVE_TDM
  __shared__ __bf16 smem[4][2][64 * 32];
  unsigned lds0 = (unsigned)(uintptr_t)(&smem[wave][0][0]);
  unsigned lds1 = (unsigned)(uintptr_t)(&smem[wave][1][0]);
  const __bf16* wbase = Wt + (size_t)n_base0 * PK;
  const int KSTEPS = PK / 32;
  tdm_load_64x32_bf16(lds0, wbase, PK);
#endif
  for (int ks = 0; ks < PK / 32; ++ks) {
    int k = ks * 32;
    H16 tb;
    int k0 = k + 8 * khalf;
#pragma unroll
    for (int half = 0; half < 2; ++half) {
      int kk = k0 + 16 * half;
      int c = kk >> 12, rem = kk & 4095, kh = rem >> 6, kw = rem & 63;
      const float* px = xb + ((size_t)c * 1280 + kh) * 960 + kw;
      f32x4 f0 = *(const f32x4*)px;
      f32x4 f1 = *(const f32x4*)(px + 4);
      int o = half * 8;
      tb.h[o + 0] = f2b(f0.x); tb.h[o + 1] = f2b(f0.y);
      tb.h[o + 2] = f2b(f0.z); tb.h[o + 3] = f2b(f0.w);
      tb.h[o + 4] = f2b(f1.x); tb.h[o + 5] = f2b(f1.y);
      tb.h[o + 6] = f2b(f1.z); tb.h[o + 7] = f2b(f1.w);
    }
    bf16x16 av = __builtin_bit_cast(bf16x16, tb);
#if HAVE_TDM
    const __bf16* buf = smem[wave][ks & 1];
    if (ks + 1 < KSTEPS) {
      tdm_load_64x32_bf16((ks & 1) ? lds0 : lds1, wbase + (k + 32), PK);
      __builtin_amdgcn_s_wait_tensorcnt(1);
    } else {
      __builtin_amdgcn_s_wait_tensorcnt(0);
    }
#pragma unroll
    for (int t = 0; t < 4; ++t)
      acc[t] = wmma_bf(av, bfrag_from_lds(buf, t, lane), acc[t]);
#else
#pragma unroll
    for (int t = 0; t < 4; ++t)
      acc[t] = wmma_bf(av, load_bfrag(Wt, PK, n_base0 + 16 * t, k, lane), acc[t]);
#endif
  }
#pragma unroll
  for (int i = 0; i < 8; ++i) {
    int row = row_base + i + 8 * khalf;
    if (row < 600) {
      int b2 = row / NPATCH, p2 = row % NPATCH;
      int tok = (p2 < MIDTOK) ? p2 : p2 + 1;
      float* orow = tokens + ((size_t)b2 * LSEQ + tok) * D_MODEL;
      const float* prow = pos + (size_t)tok * D_MODEL;
#pragma unroll
      for (int t = 0; t < 4; ++t) {
        int col = n_base0 + 16 * t + (lane & 15);
        orow[col] = acc[t][i] + pbias[col] + prow[col];
      }
    }
  }
}

__global__ void k_cls(const float* __restrict__ cls, const float* __restrict__ pos,
                      float* __restrict__ tokens) {
  int n = blockIdx.x * blockDim.x + threadIdx.x;
  if (n >= 2 * D_MODEL) return;
  int b = n / D_MODEL, c = n % D_MODEL;
  tokens[((size_t)b * LSEQ + MIDTOK) * D_MODEL + c] = cls[c] + pos[(size_t)MIDTOK * D_MODEL + c];
}

// ---------------- pre-norm: residual += hidden; hbf = bf16(rmsnorm(residual)*nw) ----
__global__ __launch_bounds__(256) void k_prenorm(
    float* __restrict__ residual, const float* __restrict__ hidden,
    const float* __restrict__ nw, __bf16* __restrict__ hbf) {
  int row = blockIdx.x;
  size_t base = (size_t)row * D_MODEL;
  __shared__ float red[8];
  float vals[3], ss = 0.f;
#pragma unroll
  for (int i = 0; i < 3; ++i) {
    int c = threadIdx.x + i * 256;
    float v = residual[base + c] + hidden[base + c];
    vals[i] = v; ss += v * v;
  }
#pragma unroll
  for (int off = 16; off; off >>= 1) ss += __shfl_down(ss, off, 32);
  if ((threadIdx.x & 31) == 0) red[threadIdx.x >> 5] = ss;
  __syncthreads();
  if (threadIdx.x == 0) {
    float t = 0.f;
#pragma unroll
    for (int i = 0; i < 8; ++i) t += red[i];
    red[0] = rsqrtf(t / (float)D_MODEL + EPSF);
  }
  __syncthreads();
  float inv = red[0];
#pragma unroll
  for (int i = 0; i < 3; ++i) {
    int c = threadIdx.x + i * 256;
    residual[base + c] = vals[i];
    hbf[base + c] = f2b(vals[i] * inv * nw[c]);
  }
}

__global__ __launch_bounds__(256) void k_finalnorm(
    const float* __restrict__ residual, const float* __restrict__ hidden,
    const float* __restrict__ nw, float* __restrict__ out) {
  int row = blockIdx.x;
  size_t base = (size_t)row * D_MODEL;
  __shared__ float red[8];
  float vals[3], ss = 0.f;
#pragma unroll
  for (int i = 0; i < 3; ++i) {
    int c = threadIdx.x + i * 256;
    float v = residual[base + c] + hidden[base + c];
    vals[i] = v; ss += v * v;
  }
#pragma unroll
  for (int off = 16; off; off >>= 1) ss += __shfl_down(ss, off, 32);
  if ((threadIdx.x & 31) == 0) red[threadIdx.x >> 5] = ss;
  __syncthreads();
  if (threadIdx.x == 0) {
    float t = 0.f;
#pragma unroll
    for (int i = 0; i < 8; ++i) t += red[i];
    red[0] = rsqrtf(t / (float)D_MODEL + EPSF);
  }
  __syncthreads();
  float inv = red[0];
#pragma unroll
  for (int i = 0; i < 3; ++i) {
    int c = threadIdx.x + i * 256;
    out[base + c] = vals[i] * inv * nw[c];
  }
}

// ---------------- xz = h @ Win^T : (602x768)x(3072x768)^T, TDM weight stream --------
__global__ __launch_bounds__(128) void k_gemm_xz(
    const __bf16* __restrict__ A, const __bf16* __restrict__ Bt, float* __restrict__ C) {
  int lane = threadIdx.x & 31, wave = threadIdx.x >> 5;
  int row_base = blockIdx.x * 16;
  int n_base0 = (blockIdx.y * 4 + wave) * 64;
  v8f acc[4];
#pragma unroll
  for (int t = 0; t < 4; ++t) zero8(acc[t]);
#if HAVE_TDM
  __shared__ __bf16 smem[4][2][64 * 32];
  unsigned lds0 = (unsigned)(uintptr_t)(&smem[wave][0][0]);
  unsigned lds1 = (unsigned)(uintptr_t)(&smem[wave][1][0]);
  const __bf16* wbase = Bt + (size_t)n_base0 * D_MODEL;
  const int KSTEPS = D_MODEL / 32;
  tdm_load_64x32_bf16(lds0, wbase, D_MODEL);
  for (int ks = 0; ks < KSTEPS; ++ks) {
    int k = ks * 32;
    const __bf16* buf = smem[wave][ks & 1];
    if (ks + 1 < KSTEPS) {
      tdm_load_64x32_bf16((ks & 1) ? lds0 : lds1, wbase + (k + 32), D_MODEL);
      __builtin_amdgcn_s_wait_tensorcnt(1);
    } else {
      __builtin_amdgcn_s_wait_tensorcnt(0);
    }
    bf16x16 av = load_afrag(A, D_MODEL, row_base, k, lane, MROWS - 1);
#pragma unroll
    for (int t = 0; t < 4; ++t)
      acc[t] = wmma_bf(av, bfrag_from_lds(buf, t, lane), acc[t]);
  }
#else
  for (int k = 0; k < D_MODEL; k += 32) {
    bf16x16 av = load_afrag(A, D_MODEL, row_base, k, lane, MROWS - 1);
#pragma unroll
    for (int t = 0; t < 4; ++t)
      acc[t] = wmma_bf(av, load_bfrag(Bt, D_MODEL, n_base0 + 16 * t, k, lane), acc[t]);
  }
#endif
  int khalf = lane >> 4;
#pragma unroll
  for (int t = 0; t < 4; ++t) {
    int col = n_base0 + 16 * t + (lane & 15);
#pragma unroll
    for (int i = 0; i < 8; ++i) {
      int row = row_base + i + 8 * khalf;
      if (row < MROWS) C[(size_t)row * D_XZ + col] = acc[t][i];
    }
  }
}

// ---------------- causal depthwise conv (k=4) + SiLU, both directions --------------
__global__ void k_conv(const float* __restrict__ xz,
                       const float* __restrict__ cwf, const float* __restrict__ cbf,
                       const float* __restrict__ cwb, const float* __restrict__ cbb,
                       float* __restrict__ uc, __bf16* __restrict__ ucbf) {
  int d = blockIdx.x * blockDim.x + threadIdx.x;
  int row = blockIdx.y;
  int dir = blockIdx.z;
  int b = row / LSEQ, t = row % LSEQ;
  const float* w = (dir ? cwb : cwf) + (size_t)d * 4;
  float acc = (dir ? cbb : cbf)[d];
#pragma unroll
  for (int j = 0; j < 4; ++j) {
    int i = t - 3 + j;
    if (i >= 0) {
      int ts = dir ? (LSEQ - 1 - i) : i;
      acc += w[j] * xz[((size_t)b * LSEQ + ts) * D_XZ + d];
    }
  }
  float s = silu(acc);
  size_t o = (size_t)dir * MROWS * D_INNER + ((size_t)b * LSEQ + t) * D_INNER + d;
  uc[o] = s;
  ucbf[o] = f2b(s);
}

// ---------------- x_dbl = uc @ Wx^T : (602x1536)x(80x1536)^T, both dirs -------------
__global__ __launch_bounds__(128) void k_gemm_xdbl(
    const __bf16* __restrict__ ucbf, const __bf16* __restrict__ wx_f,
    const __bf16* __restrict__ wx_b, float* __restrict__ xdbl, __bf16* __restrict__ dtbf) {
  int dir = blockIdx.z;
  const __bf16* A = ucbf + (size_t)dir * MROWS * D_INNER;
  const __bf16* Bt = dir ? wx_b : wx_f;
  float* xo = xdbl + (size_t)dir * MROWS * XDBL_N;
  __bf16* dto = dtbf + (size_t)dir * MROWS * DT_PAD;

  int lane = threadIdx.x & 31, wave = threadIdx.x >> 5;
  int row_base = blockIdx.x * 16;
  int n_base = (blockIdx.y * 4 + wave) * 16;
  if (n_base >= XDBL_N) return;

  v8f acc; zero8(acc);
  for (int k = 0; k < D_INNER; k += 32) {
    bf16x16 av = load_afrag(A, D_INNER, row_base, k, lane, MROWS - 1);
    acc = wmma_bf(av, load_bfrag(Bt, D_INNER, n_base, k, lane), acc);
  }
  int khalf = lane >> 4;
  int col = n_base + (lane & 15);
#pragma unroll
  for (int i = 0; i < 8; ++i) {
    int row = row_base + i + 8 * khalf;
    if (row < MROWS) {
      xo[(size_t)row * XDBL_N + col] = acc[i];
      if (col < DT_RANK) dto[(size_t)row * DT_PAD + col] = f2b(acc[i]);
    }
  }
}

// ---------------- delta = softplus(dt @ Wdt^T + bdt), both dirs ----------------
__global__ __launch_bounds__(128) void k_gemm_delta(
    const __bf16* __restrict__ dtbf, const __bf16* __restrict__ wdt_f,
    const __bf16* __restrict__ wdt_b, const float* __restrict__ bdt_f,
    const float* __restrict__ bdt_b, float* __restrict__ delta) {
  int dir = blockIdx.z;
  const __bf16* A = dtbf + (size_t)dir * MROWS * DT_PAD;
  const __bf16* Bt = dir ? wdt_b : wdt_f;
  const float* bdt = dir ? bdt_b : bdt_f;
  float* dout = delta + (size_t)dir * MROWS * D_INNER;

  int lane = threadIdx.x & 31, wave = threadIdx.x >> 5;
  int row_base = blockIdx.x * 16;
  int n_base0 = (blockIdx.y * 4 + wave) * 64;
  v8f acc[4];
#pragma unroll
  for (int t = 0; t < 4; ++t) zero8(acc[t]);
  for (int k = 0; k < DT_PAD; k += 32) {
    bf16x16 av = load_afrag(A, DT_PAD, row_base, k, lane, MROWS - 1);
#pragma unroll
    for (int t = 0; t < 4; ++t)
      acc[t] = wmma_bf(av, load_bfrag(Bt, DT_PAD, n_base0 + 16 * t, k, lane), acc[t]);
  }
  int khalf = lane >> 4;
#pragma unroll
  for (int t = 0; t < 4; ++t) {
    int col = n_base0 + 16 * t + (lane & 15);
#pragma unroll
    for (int i = 0; i < 8; ++i) {
      int row = row_base + i + 8 * khalf;
      if (row < MROWS) {
        float v = acc[t][i] + bdt[col];
        dout[(size_t)row * D_INNER + col] = (v > 20.f) ? v : log1pf(__expf(v));
      }
    }
  }
}

// ---------------- selective scan: 1 lane per (b,d), 16 states in regs --------------
__global__ __launch_bounds__(256) void k_scan(
    const float* __restrict__ delta, const float* __restrict__ xdbl,
    const float* __restrict__ uc,
    const float* __restrict__ Alog_f, const float* __restrict__ Alog_b,
    const float* __restrict__ Dsk_f, const float* __restrict__ Dsk_b,
    float* __restrict__ ydir) {
  int dir = blockIdx.z, b = blockIdx.y;
  int d = blockIdx.x * blockDim.x + threadIdx.x;
  const float* Al = dir ? Alog_b : Alog_f;
  float Dv = (dir ? Dsk_b : Dsk_f)[d];
  size_t NIN = (size_t)MROWS * D_INNER, NDB = (size_t)MROWS * XDBL_N;
  const float* dl = delta + (size_t)dir * NIN;
  const float* xb = xdbl + (size_t)dir * NDB;
  const float* ub = uc + (size_t)dir * NIN;
  float* yo = ydir + (size_t)dir * NIN;

  float Ar[D_STATE], h[D_STATE];
#pragma unroll
  for (int s = 0; s < D_STATE; ++s) { Ar[s] = -__expf(Al[(size_t)d * D_STATE + s]); h[s] = 0.f; }

  __shared__ float BC[32];
  for (int t = 0; t < LSEQ; ++t) {
    size_t rb = (size_t)b * LSEQ + t;
    if (threadIdx.x < 32) BC[threadIdx.x] = xb[rb * XDBL_N + DT_RANK + threadIdx.x];
    __syncthreads();
    float dt = dl[rb * D_INNER + d];
    float u  = ub[rb * D_INNER + d];
    float y = 0.f;
#pragma unroll
    for (int s = 0; s < D_STATE; ++s) {
      float dA = __expf(dt * Ar[s]);
      h[s] = dA * h[s] + (dt * BC[s]) * u;
      y += h[s] * BC[D_STATE + s];
    }
    yo[rb * D_INNER + d] = y + Dv * u;
    __syncthreads();
  }
}

// ---------------- gate: y = (yf + reverse(yb)) * silu(z), bf16 out ----------------
__global__ void k_gate(const float* __restrict__ ydir, const float* __restrict__ xz,
                       __bf16* __restrict__ ybf) {
  int d = blockIdx.x * blockDim.x + threadIdx.x;
  int row = blockIdx.y;
  int b = row / LSEQ, t = row % LSEQ;
  size_t NIN = (size_t)MROWS * D_INNER;
  float z = xz[((size_t)b * LSEQ + t) * D_XZ + D_INNER + d];
  float v = ydir[((size_t)b * LSEQ + t) * D_INNER + d]
          + ydir[NIN + ((size_t)b * LSEQ + (LSEQ - 1 - t)) * D_INNER + d];
  ybf[((size_t)b * LSEQ + t) * D_INNER + d] = f2b(v * silu(z));
}

// ---------------- hidden = (y @ Wout^T)/2, TDM weight stream ----------------
__global__ __launch_bounds__(128) void k_gemm_out(
    const __bf16* __restrict__ A, const __bf16* __restrict__ Bt, float* __restrict__ C) {
  int lane = threadIdx.x & 31, wave = threadIdx.x >> 5;
  int row_base = blockIdx.x * 16;
  int n_base0 = (blockIdx.y * 4 + wave) * 64;
  v8f acc[4];
#pragma unroll
  for (int t = 0; t < 4; ++t) zero8(acc[t]);
#if HAVE_TDM
  __shared__ __bf16 smem[4][2][64 * 32];
  unsigned lds0 = (unsigned)(uintptr_t)(&smem[wave][0][0]);
  unsigned lds1 = (unsigned)(uintptr_t)(&smem[wave][1][0]);
  const __bf16* wbase = Bt + (size_t)n_base0 * D_INNER;
  const int KSTEPS = D_INNER / 32;
  tdm_load_64x32_bf16(lds0, wbase, D_INNER);
  for (int ks = 0; ks < KSTEPS; ++ks) {
    int k = ks * 32;
    const __bf16* buf = smem[wave][ks & 1];
    if (ks + 1 < KSTEPS) {
      tdm_load_64x32_bf16((ks & 1) ? lds0 : lds1, wbase + (k + 32), D_INNER);
      __builtin_amdgcn_s_wait_tensorcnt(1);
    } else {
      __builtin_amdgcn_s_wait_tensorcnt(0);
    }
    bf16x16 av = load_afrag(A, D_INNER, row_base, k, lane, MROWS - 1);
#pragma unroll
    for (int t = 0; t < 4; ++t)
      acc[t] = wmma_bf(av, bfrag_from_lds(buf, t, lane), acc[t]);
  }
#else
  for (int k = 0; k < D_INNER; k += 32) {
    bf16x16 av = load_afrag(A, D_INNER, row_base, k, lane, MROWS - 1);
#pragma unroll
    for (int t = 0; t < 4; ++t)
      acc[t] = wmma_bf(av, load_bfrag(Bt, D_INNER, n_base0 + 16 * t, k, lane), acc[t]);
  }
#endif
  int khalf = lane >> 4;
#pragma unroll
  for (int t = 0; t < 4; ++t) {
    int col = n_base0 + 16 * t + (lane & 15);
#pragma unroll
    for (int i = 0; i < 8; ++i) {
      int row = row_base + i + 8 * khalf;
      if (row < MROWS) C[(size_t)row * D_MODEL + col] = 0.5f * acc[t][i];
    }
  }
}

// ---------------- host ----------------
extern "C" void kernel_launch(void* const* d_in, const int* in_sizes, int n_in,
                              void* d_out, int out_size, void* d_ws, size_t ws_size,
                              hipStream_t stream) {
  (void)in_sizes; (void)n_in; (void)out_size; (void)ws_size;
  const float* x       = (const float*)d_in[0];
  const float* patch_w = (const float*)d_in[1];
  const float* patch_b = (const float*)d_in[2];
  const float* cls_tok = (const float*)d_in[3];
  const float* pos     = (const float*)d_in[4];
  const float* norm_ws = (const float*)d_in[5];
  const float* Win     = (const float*)d_in[6];
  const float* convw_f = (const float*)d_in[7];
  const float* convb_f = (const float*)d_in[8];
  const float* Wx_f    = (const float*)d_in[9];
  const float* Wdt_f   = (const float*)d_in[10];
  const float* bdt_f   = (const float*)d_in[11];
  const float* Alog_f  = (const float*)d_in[12];
  const float* Dskip_f = (const float*)d_in[13];
  const float* convw_b = (const float*)d_in[14];
  const float* convb_b = (const float*)d_in[15];
  const float* Wx_b    = (const float*)d_in[16];
  const float* Wdt_b   = (const float*)d_in[17];
  const float* bdt_b   = (const float*)d_in[18];
  const float* Alog_b  = (const float*)d_in[19];
  const float* Dskip_b = (const float*)d_in[20];
  const float* Wout    = (const float*)d_in[21];
  const float* normf_w = (const float*)d_in[22];
  float* out = (float*)d_out;

  char* ws = (char*)d_ws;
  size_t off = 0;
  auto alloc = [&](size_t bytes) { size_t o = off; off += (bytes + 255) & ~(size_t)255; return o; };

  const size_t NW_PATCH = (size_t)D_MODEL * PK;
  const size_t NW_IN    = (size_t)DEPTH * D_XZ * D_MODEL;
  const size_t NW_X     = (size_t)DEPTH * XDBL_N * D_INNER;
  const size_t NW_DT    = (size_t)DEPTH * D_INNER * DT_PAD;
  const size_t NW_OUT   = (size_t)DEPTH * D_MODEL * D_INNER;
  const size_t NTOK = (size_t)MROWS * D_MODEL;
  const size_t NXZ  = (size_t)MROWS * D_XZ;
  const size_t NIN  = (size_t)MROWS * D_INNER;
  const size_t NDBL = (size_t)MROWS * XDBL_N;
  const size_t NDT  = (size_t)MROWS * DT_PAD;

  __bf16* wpatch   = (__bf16*)(ws + alloc(NW_PATCH * 2));
  __bf16* win      = (__bf16*)(ws + alloc(NW_IN * 2));
  __bf16* wx       = (__bf16*)(ws + alloc(2 * NW_X * 2));
  __bf16* wdtp     = (__bf16*)(ws + alloc(2 * NW_DT * 2));
  __bf16* wout     = (__bf16*)(ws + alloc(NW_OUT * 2));
  float*  hidden   = (float*)(ws + alloc(NTOK * 4));
  float*  residual = (float*)(ws + alloc(NTOK * 4));
  __bf16* hbf      = (__bf16*)(ws + alloc(NTOK * 2));
  float*  xz       = (float*)(ws + alloc(NXZ * 4));
  float*  uc       = (float*)(ws + alloc(2 * NIN * 4));
  __bf16* ucbf     = (__bf16*)(ws + alloc(2 * NIN * 2));
  float*  xdbl     = (float*)(ws + alloc(2 * NDBL * 4));
  __bf16* dtbf     = (__bf16*)(ws + alloc(2 * NDT * 2));
  float*  deltab   = (float*)(ws + alloc(2 * NIN * 4));
  float*  ydir     = (float*)(ws + alloc(2 * NIN * 4));
  __bf16* ybf      = (__bf16*)(ws + alloc(NIN * 2));

  auto blks = [](long n) { return (unsigned)((n + 255) / 256); };

  k_zero_f32<<<blks((long)NTOK), 256, 0, stream>>>(residual, (long)NTOK);
  k_zero_bf16<<<blks((long)(2 * NDT)), 256, 0, stream>>>(dtbf, (long)(2 * NDT));

  k_cvt<<<blks((long)NW_PATCH), 256, 0, stream>>>(patch_w, wpatch, (long)NW_PATCH);
  k_cvt<<<blks((long)NW_IN), 256, 0, stream>>>(Win, win, (long)NW_IN);
  k_cvt<<<blks((long)NW_X), 256, 0, stream>>>(Wx_f, wx, (long)NW_X);
  k_cvt<<<blks((long)NW_X), 256, 0, stream>>>(Wx_b, wx + NW_X, (long)NW_X);
  k_cvt_pad<<<blks((long)NW_DT), 256, 0, stream>>>(Wdt_f, wdtp, DEPTH * D_INNER, DT_RANK, DT_PAD);
  k_cvt_pad<<<blks((long)NW_DT), 256, 0, stream>>>(Wdt_b, wdtp + NW_DT, DEPTH * D_INNER, DT_RANK, DT_PAD);
  k_cvt<<<blks((long)NW_OUT), 256, 0, stream>>>(Wout, wout, (long)NW_OUT);

  k_patch_gemm<<<dim3(38, 3), 128, 0, stream>>>(x, wpatch, patch_b, pos, hidden);
  k_cls<<<6, 256, 0, stream>>>(cls_tok, pos, hidden);

  for (int l = 0; l < DEPTH; ++l) {
    k_prenorm<<<MROWS, 256, 0, stream>>>(residual, hidden, norm_ws + (size_t)l * D_MODEL, hbf);
    k_gemm_xz<<<dim3(38, 12), 128, 0, stream>>>(hbf, win + (size_t)l * D_XZ * D_MODEL, xz);
    k_conv<<<dim3(6, MROWS, 2), 256, 0, stream>>>(xz,
        convw_f + (size_t)l * D_INNER * 4, convb_f + (size_t)l * D_INNER,
        convw_b + (size_t)l * D_INNER * 4, convb_b + (size_t)l * D_INNER, uc, ucbf);
    k_gemm_xdbl<<<dim3(38, 2, 2), 128, 0, stream>>>(ucbf,
        wx + (size_t)l * XDBL_N * D_INNER, wx + NW_X + (size_t)l * XDBL_N * D_INNER,
        xdbl, dtbf);
    k_gemm_delta<<<dim3(38, 6, 2), 128, 0, stream>>>(dtbf,
        wdtp + (size_t)l * D_INNER * DT_PAD, wdtp + NW_DT + (size_t)l * D_INNER * DT_PAD,
        bdt_f + (size_t)l * D_INNER, bdt_b + (size_t)l * D_INNER, deltab);
    k_scan<<<dim3(6, 2, 2), 256, 0, stream>>>(deltab, xdbl, uc,
        Alog_f + (size_t)l * D_INNER * D_STATE, Alog_b + (size_t)l * D_INNER * D_STATE,
        Dskip_f + (size_t)l * D_INNER, Dskip_b + (size_t)l * D_INNER, ydir);
    k_gate<<<dim3(6, MROWS), 256, 0, stream>>>(ydir, xz, ybf);
    k_gemm_out<<<dim3(38, 3), 128, 0, stream>>>(ybf, wout + (size_t)l * D_MODEL * D_INNER, hidden);
  }

  k_finalnorm<<<MROWS, 256, 0, stream>>>(residual, hidden, normf_w, out);
}
